// AxialGenerator_51436528337513
// MI455X (gfx1250) — matrix operand order repaired and checked by reading, not compiled, not run on hardware
//
#include <hip/hip_runtime.h>
#include <hip/hip_bf16.h>

typedef __attribute__((ext_vector_type(16))) __bf16 v16bf;
typedef __attribute__((ext_vector_type(8)))  float  v8f;

#define EMB   128
#define HW    4096
#define MROWS 8192      // b*s*c = 16*64*8

// ---------------------------------------------------------------------------
// Weight packing into WMMA B-fragment order (bf16).
// B-frag layout (16-bit, 32x16 K x N tile):
//   lane 0-15  -> N = lane,    K = {0..7, 16..23}   (e = 0..15)
//   lane 16-31 -> N = lane-16, K = {8..15, 24..31}
// Fragment storage: [frag][lane][e] contiguous, 32 bytes/lane -> b128 pairs.
// ---------------------------------------------------------------------------
__global__ __launch_bounds__(256)
void axgen_pack_enc(const float* __restrict__ W /*(128,4096)*/, __bf16* __restrict__ Bp) {
    int idx = blockIdx.x * 256 + threadIdx.x;          // 128kt * 8nt * 32lane * 16e
    int e    = idx & 15;
    int lane = (idx >> 4) & 31;
    int f    = idx >> 9;
    int nt   = f & 7;
    int kt   = f >> 3;
    int n = nt * 16 + (lane & 15);
    int k = kt * 32 + ((lane >= 16) ? 8 : 0) + ((e < 8) ? e : e + 8);
    Bp[idx] = (__bf16)W[(size_t)n * HW + k];           // B[k][n] = enc_w[n][k]
}

__global__ __launch_bounds__(256)
void axgen_pack_dec(const float* __restrict__ W /*(4096,128)*/, __bf16* __restrict__ Bp) {
    int idx = blockIdx.x * 256 + threadIdx.x;          // 4kt * 256nt * 32lane * 16e
    int e    = idx & 15;
    int lane = (idx >> 4) & 31;
    int f    = idx >> 9;
    int nt   = f & 255;
    int kt   = f >> 8;
    int n = nt * 16 + (lane & 15);
    int k = kt * 32 + ((lane >= 16) ? 8 : 0) + ((e < 8) ? e : e + 8);
    Bp[idx] = (__bf16)W[(size_t)n * EMB + k];          // B[k][n] = dec_w[n][k]
}

// ---------------------------------------------------------------------------
// A-fragment builder: 16x32 bf16 tile per wave, per ISA 16-bit A layout.
//   lanes 0-15:  M=lane,    K = {0..7, 16..23}
//   lanes 16-31: M=lane-16, K = {8..15, 24..31}
// src points at row start + kTile*32 + (hi?8:0); stride handled by caller.
// ---------------------------------------------------------------------------
__device__ __forceinline__ v16bf axgen_load_afrag(const float* __restrict__ ap) {
    float4 f0 = *(const float4*)(ap);
    float4 f1 = *(const float4*)(ap + 4);
    float4 f2 = *(const float4*)(ap + 16);
    float4 f3 = *(const float4*)(ap + 20);
    v16bf a;
    a[0]=(__bf16)f0.x;  a[1]=(__bf16)f0.y;  a[2]=(__bf16)f0.z;  a[3]=(__bf16)f0.w;
    a[4]=(__bf16)f1.x;  a[5]=(__bf16)f1.y;  a[6]=(__bf16)f1.z;  a[7]=(__bf16)f1.w;
    a[8]=(__bf16)f2.x;  a[9]=(__bf16)f2.y;  a[10]=(__bf16)f2.z; a[11]=(__bf16)f2.w;
    a[12]=(__bf16)f3.x; a[13]=(__bf16)f3.y; a[14]=(__bf16)f3.z; a[15]=(__bf16)f3.w;
    return a;
}

// ---------------------------------------------------------------------------
// Encoder: E(8192x128) = X(8192x4096) * enc_w^T + enc_b      (bf16 WMMA, f32 acc)
// Block = 256 thr (8 waves); wave handles a 16x128 strip (8 N-tiles).
// ---------------------------------------------------------------------------
__global__ __launch_bounds__(256)
void axgen_enc_gemm(const float* __restrict__ X, const __bf16* __restrict__ Bp,
                    const float* __restrict__ bias, float* __restrict__ E) {
    const int wave = threadIdx.x >> 5;
    const int lane = threadIdx.x & 31;
    const int mBase = (blockIdx.x * 8 + wave) * 16;
    const bool hi = lane >= 16;
    const int row = mBase + (lane & 15);
    const float* ap_row = X + (size_t)row * HW + (hi ? 8 : 0);

    v8f acc[8] = {};
    for (int kt = 0; kt < 128; ++kt) {
        const float* ap = ap_row + kt * 32;
        __builtin_prefetch(ap + 8 * 32, 0, 1);         // global_prefetch_b8 ahead in K
        v16bf a = axgen_load_afrag(ap);
        #pragma unroll
        for (int nt = 0; nt < 8; ++nt) {
            v16bf b = *(const v16bf*)(Bp + (((size_t)(kt * 8 + nt) * 32 + lane) << 4));
            acc[nt] = __builtin_amdgcn_wmma_f32_16x16x32_bf16(
                false, a, false, b, (short)0, acc[nt], false, false);
        }
    }
    // C layout: lanes 0-15 -> M=r, N=lane ; lanes 16-31 -> M=r+8, N=lane-16
    #pragma unroll
    for (int nt = 0; nt < 8; ++nt) {
        int col = nt * 16 + (lane & 15);
        float bv = bias[col];
        #pragma unroll
        for (int r = 0; r < 8; ++r) {
            int m = mBase + r + (hi ? 8 : 0);
            E[(size_t)m * EMB + col] = acc[nt][r] + bv;
        }
    }
}

// ---------------------------------------------------------------------------
// Decoder: Out(8192x4096) = E(8192x128) * dec_w^T + dec_b    (bf16 WMMA)
// Grid = 64 (M/128) x 32 (N/128); wave -> 16x128 strip.
// ---------------------------------------------------------------------------
__global__ __launch_bounds__(256)
void axgen_dec_gemm(const float* __restrict__ E, const __bf16* __restrict__ Bp,
                    const float* __restrict__ bias, float* __restrict__ Out) {
    const int wave = threadIdx.x >> 5;
    const int lane = threadIdx.x & 31;
    const int mb = blockIdx.x & 63;
    const int nb = blockIdx.x >> 6;
    const int mBase = mb * 128 + wave * 16;
    const bool hi = lane >= 16;
    const int row = mBase + (lane & 15);
    const float* ap_row = E + (size_t)row * EMB + (hi ? 8 : 0);

    v8f acc[8] = {};
    #pragma unroll
    for (int kt = 0; kt < 4; ++kt) {
        v16bf a = axgen_load_afrag(ap_row + kt * 32);
        #pragma unroll
        for (int nt = 0; nt < 8; ++nt) {
            size_t f = (size_t)kt * 256 + nb * 8 + nt;
            v16bf b = *(const v16bf*)(Bp + ((f * 32 + lane) << 4));
            acc[nt] = __builtin_amdgcn_wmma_f32_16x16x32_bf16(
                false, a, false, b, (short)0, acc[nt], false, false);
        }
    }
    #pragma unroll
    for (int nt = 0; nt < 8; ++nt) {
        int col = nb * 128 + nt * 16 + (lane & 15);
        float bv = bias[col];
        #pragma unroll
        for (int r = 0; r < 8; ++r) {
            int m = mBase + r + (hi ? 8 : 0);
            Out[(size_t)m * HW + col] = acc[nt][r] + bv;
        }
    }
}

// ---------------------------------------------------------------------------
// Axial self-attention over one sequence per workgroup (L = 64 or 8, D=128).
// HEADS=8, DH=4, scale = DH^-0.5 = 0.5. Online softmax along j.
// mode 0: write projection into outBuf (the out_s temp)
// mode 1: out_c + tmpIn, leaky_relu(0.2), write into outBuf (next e)
// ---------------------------------------------------------------------------
__global__ __launch_bounds__(256)
void axgen_attn(const float* __restrict__ eIn, float* __restrict__ outBuf,
                const float* __restrict__ tmpIn,
                const float* __restrict__ wq,  const float* __restrict__ wkv,
                const float* __restrict__ wout,const float* __restrict__ bout,
                int L, int tokenStride, int div, int strideA, int strideB, int mode) {
    __shared__ float smem[14336];                 // 56 KB of the 320 KB WGP pool
    float* sE = smem;                             // L*128   (<= 8192)
    float* sQ = smem + 8192;                      // L*32
    float* sK = smem + 8192 + 2048;
    float* sV = smem + 8192 + 4096;
    float* sO = smem;                             // reuse E region after QKV

    const int tid = threadIdx.x;
    const int seq = blockIdx.x;
    const size_t base = (size_t)(seq / div) * strideA + (size_t)(seq % div) * strideB;

    for (int i = tid; i < L * 128; i += 256) {
        int tok = i >> 7, d = i & 127;
        sE[i] = eIn[base + (size_t)tok * tokenStride + d];
    }
    __syncthreads();

    for (int t = tid; t < L * 96; t += 256) {     // fused q | k | v projections
        int tok = t / 96, j = t - tok * 96;
        const float* wrow = (j < 32) ? (wq + j * 128) : (wkv + (j - 32) * 128);
        const float* er = sE + tok * 128;
        float acc = 0.f;
        #pragma unroll 8
        for (int d = 0; d < 128; ++d) acc += er[d] * wrow[d];
        if      (j < 32) sQ[tok * 32 + j]      = acc;
        else if (j < 64) sK[tok * 32 + j - 32] = acc;
        else             sV[tok * 32 + j - 64] = acc;
    }
    __syncthreads();

    for (int t = tid; t < 8 * L; t += 256) {      // per (head, query-row)
        int h = t & 7, i = t >> 3;
        const float* qr = sQ + i * 32 + h * 4;
        float q0 = qr[0], q1 = qr[1], q2 = qr[2], q3 = qr[3];
        float m = -1e30f, s = 0.f, o0 = 0.f, o1 = 0.f, o2 = 0.f, o3 = 0.f;
        for (int j = 0; j < L; ++j) {
            const float* kr = sK + j * 32 + h * 4;
            float d = (q0 * kr[0] + q1 * kr[1] + q2 * kr[2] + q3 * kr[3]) * 0.5f;
            float nm = fmaxf(m, d);
            float c = __expf(m - nm);
            float p = __expf(d - nm);
            const float* vr = sV + j * 32 + h * 4;
            s  = s  * c + p;
            o0 = o0 * c + p * vr[0];
            o1 = o1 * c + p * vr[1];
            o2 = o2 * c + p * vr[2];
            o3 = o3 * c + p * vr[3];
            m = nm;
        }
        float inv = 1.f / s;
        float* orow = sO + i * 32 + h * 4;
        orow[0] = o0 * inv; orow[1] = o1 * inv; orow[2] = o2 * inv; orow[3] = o3 * inv;
    }
    __syncthreads();

    for (int t = tid; t < L * 128; t += 256) {    // output projection + combine
        int tok = t >> 7, d = t & 127;
        const float* orow = sO + tok * 32;
        const float* wrow = wout + d * 32;
        float acc = bout[d];
        #pragma unroll
        for (int k = 0; k < 32; ++k) acc += orow[k] * wrow[k];
        size_t gi = base + (size_t)tok * tokenStride + d;
        if (mode == 0) {
            outBuf[gi] = acc;
        } else {
            float r = acc + tmpIn[gi];
            outBuf[gi] = (r > 0.f) ? r : 0.2f * r; // leaky_relu(0.2)
        }
    }
}

// ---------------------------------------------------------------------------
extern "C" void kernel_launch(void* const* d_in, const int* in_sizes, int n_in,
                              void* d_out, int out_size, void* d_ws, size_t ws_size,
                              hipStream_t stream) {
    const float* x     = (const float*)d_in[0];
    const float* enc_w = (const float*)d_in[1];
    const float* enc_b = (const float*)d_in[2];
    const float* dec_w = (const float*)d_in[3];
    const float* dec_b = (const float*)d_in[4];
    const float* wq    = (const float*)d_in[5];   // (3,2,32,128)
    const float* wkv   = (const float*)d_in[6];   // (3,2,64,128)
    const float* wout  = (const float*)d_in[7];   // (3,2,128,32)
    const float* bout  = (const float*)d_in[8];   // (3,2,128)
    float* out = (float*)d_out;

    char* ws = (char*)d_ws;                       // ~14 MB used
    float*  eA  = (float*)(ws);                   // 4 MB
    float*  eB  = (float*)(ws + (4  << 20));      // 4 MB
    float*  tmp = (float*)(ws + (8  << 20));      // 4 MB (out_s)
    __bf16* BpE = (__bf16*)(ws + (12 << 20));     // 1 MB packed enc_w
    __bf16* BpD = (__bf16*)(ws + (13 << 20));     // 1 MB packed dec_w

    axgen_pack_enc<<<2048, 256, 0, stream>>>(enc_w, BpE);
    axgen_pack_dec<<<2048, 256, 0, stream>>>(dec_w, BpD);

    axgen_enc_gemm<<<MROWS / 128, 256, 0, stream>>>(x, BpE, enc_b, eA);

    float* cur = eA;
    float* nxt = eB;
    for (int l = 0; l < 3; ++l) {
        const float* wq0 = wq  + (size_t)(l * 2) * 4096;
        const float* wk0 = wkv + (size_t)(l * 2) * 8192;
        const float* wo0 = wout+ (size_t)(l * 2) * 4096;
        const float* bo0 = bout+ (size_t)(l * 2) * 128;
        // axis s: 128 sequences of L=64, token stride = c*EMB = 1024
        axgen_attn<<<128, 256, 0, stream>>>(cur, tmp, tmp, wq0, wk0, wo0, bo0,
                                            64, 1024, 8, 65536, 128, 0);
        // axis c: 1024 sequences of L=8, token stride = EMB = 128; combine+LReLU
        axgen_attn<<<1024, 256, 0, stream>>>(cur, nxt, tmp,
                                             wq0 + 4096, wk0 + 8192, wo0 + 4096, bo0 + 128,
                                             8, 128, 64, 65536, 1024, 1);
        float* t2 = cur; cur = nxt; nxt = t2;
    }

    axgen_dec_gemm<<<64 * 32, 256, 0, stream>>>(cur, BpD, dec_b, out);
}
